// Decoder_Block_74174085202568
// MI455X (gfx1250) — matrix operand order repaired and checked by reading, not compiled, not run on hardware
//
#include <hip/hip_runtime.h>

// ---------------------------------------------------------------------------
// CDNA5 / gfx1250 implementation of the Decoder_Block reference.
// wave32, WMMA 16x16x32 bf16 (fp32 accumulate). HBM-bound workload:
//   big tensors (8,128,128,256) stay fp32 in HBM; A tiles DMA'd into LDS by
//   the Tensor Data Mover (TENSOR_LOAD_TO_LDS + s_wait_tensorcnt), converted
//   to bf16 fragments in LDS; 16 weight matrices pre-swizzled to bf16
//   fragment order (2 MB, L2-resident).
// ---------------------------------------------------------------------------

typedef __bf16 bf16;
typedef __bf16 v16bf __attribute__((ext_vector_type(16)));
typedef __bf16 v8bf  __attribute__((ext_vector_type(8)));
typedef float  v8f   __attribute__((ext_vector_type(8)));
typedef unsigned int v4u __attribute__((ext_vector_type(4)));
typedef int v8i __attribute__((ext_vector_type(8)));
typedef int v4i __attribute__((ext_vector_type(4)));

#define DIMC  256
#define NSEQ  128
#define BATCH 8
#define RBIG  (BATCH * NSEQ * NSEQ)   // 131072 rows of the edge tensors
#define RSML  (BATCH * NSEQ)          // 1024 rows of the node tensors
#define LN_EPSF 1e-5f

#if __has_builtin(__builtin_amdgcn_tensor_load_to_lds) && \
    __has_builtin(__builtin_amdgcn_s_wait_tensorcnt)
#define USE_TDM 1
#else
#define USE_TDM 0
#endif

__device__ __forceinline__ unsigned pack_bf16x2(float a, float b) {
  unsigned short x = __builtin_bit_cast(unsigned short, (__bf16)a);
  unsigned short y = __builtin_bit_cast(unsigned short, (__bf16)b);
  return (unsigned)x | ((unsigned)y << 16);
}

// ---------------------------------------------------------------------------
// Weight swizzle: fp32 W[k][n] (256x256) -> bf16 fragment order so that each
// lane's B-fragment for v_wmma_f32_16x16x32_bf16 is one contiguous 32B load.
// Layout index: ((nt*8 + kt)*32 + lane)*16 + j, with
//   n = nt*16 + (lane&15),  k = kt*32 + (lane>>4)*16 + j   (j = 0..15)
// ---------------------------------------------------------------------------
__global__ __launch_bounds__(256) void swizzle_weights(
    const float* __restrict__ W, bf16* __restrict__ out) {
  int t    = blockIdx.x * 256 + threadIdx.x;   // 0..65535
  int j    = t & 15;
  int L    = (t >> 4) & 31;
  int kt   = (t >> 9) & 7;
  int nt   = t >> 12;
  int half = L >> 4;
  int n = nt * 16 + (L & 15);
  int k = kt * 32 + half * 16 + j;
  out[t] = (bf16)W[k * DIMC + n];
}

// ---------------------------------------------------------------------------
// LayerNorm over last dim (256). One wave32 per row, 8 rows per block.
// ---------------------------------------------------------------------------
__global__ __launch_bounds__(256) void ln_kernel(
    const float* __restrict__ x, const float* __restrict__ g,
    const float* __restrict__ bta, float* __restrict__ y) {
  const int wave = threadIdx.x >> 5;
  const int lane = threadIdx.x & 31;
  const size_t row = (size_t)blockIdx.x * 8 + wave;
  const float* xr = x + row * DIMC;
  float4 v0 = ((const float4*)xr)[lane];        // cols 4*lane .. 4*lane+3
  float4 v1 = ((const float4*)xr)[32 + lane];   // cols 128+4*lane ..
  float s  = v0.x + v0.y + v0.z + v0.w + v1.x + v1.y + v1.z + v1.w;
  float ss = v0.x*v0.x + v0.y*v0.y + v0.z*v0.z + v0.w*v0.w
           + v1.x*v1.x + v1.y*v1.y + v1.z*v1.z + v1.w*v1.w;
#pragma unroll
  for (int off = 16; off > 0; off >>= 1) {
    s  += __shfl_xor(s, off);
    ss += __shfl_xor(ss, off);
  }
  float mu  = s * (1.0f / DIMC);
  float var = ss * (1.0f / DIMC) - mu * mu;
  float inv = rsqrtf(var + LN_EPSF);
  float4 g0 = ((const float4*)g)[lane],   g1 = ((const float4*)g)[32 + lane];
  float4 b0 = ((const float4*)bta)[lane], b1 = ((const float4*)bta)[32 + lane];
  float* yr = y + row * DIMC;
  float4 o0, o1;
  o0.x = (v0.x - mu) * inv * g0.x + b0.x;  o0.y = (v0.y - mu) * inv * g0.y + b0.y;
  o0.z = (v0.z - mu) * inv * g0.z + b0.z;  o0.w = (v0.w - mu) * inv * g0.w + b0.w;
  o1.x = (v1.x - mu) * inv * g1.x + b1.x;  o1.y = (v1.y - mu) * inv * g1.y + b1.y;
  o1.z = (v1.z - mu) * inv * g1.z + b1.z;  o1.w = (v1.w - mu) * inv * g1.w + b1.w;
  ((float4*)yr)[lane]      = o0;
  ((float4*)yr)[32 + lane] = o1;
}

// ---------------------------------------------------------------------------
// GEMM: C[r,n] = act( sum_k A[r,k] * W[k,n] + bias[n] + res[r,n] )
// A fp32 row-major [rows,256]; W pre-swizzled bf16; C fp32.
// Block = 256 threads = 8 waves; tile 64 rows x 256 cols; K unrolled x8.
// A tile staged via TDM (tensor_load_to_lds) -> fp32 LDS -> bf16 LDS tile.
// Each wave: 64x32 slab -> acc[4][2] of v_wmma_f32_16x16x32_bf16.
// ---------------------------------------------------------------------------
__global__ __launch_bounds__(256) void gemm256(
    const float* __restrict__ A, const bf16* __restrict__ Wsw,
    const float* __restrict__ bias, const float* __restrict__ res,
    float* __restrict__ C, int relu) {
#if USE_TDM
  __shared__ float    Afp[64 * DIMC];      // 64 KB fp32 tile (TDM target)
#endif
  __shared__ unsigned As[64 * DIMC / 2];   // 64x256 bf16 tile, 32 KB

  const int tid  = threadIdx.x;
  const int wave = tid >> 5;
  const int lane = tid & 31;
  const int half = lane >> 4;
  const int l15  = lane & 15;
  const size_t row0 = (size_t)blockIdx.x * 64;

#if USE_TDM
  // --- Tensor Data Mover: DMA 64x256 fp32 tile Global -> LDS -------------
  // D# group0: count=1 | lds_addr | global_addr | type=2 ("image")
  // D# group1: data_size=4B, tensor_dim0=256, tensor_dim1=64,
  //            tile_dim0=256, tile_dim1=64, tensor_dim0_stride=256
  if (wave == 0) {                       // wave-uniform branch: 1 DMA / block
    unsigned lds = (unsigned)(size_t)(void*)Afp;
    unsigned long long ga = (unsigned long long)(size_t)(A + row0 * DIMC);
    v4u g0;
    g0[0] = 1u;                                      // count=1 (user D#)
    g0[1] = lds;                                     // lds_addr (bytes)
    g0[2] = (unsigned)ga;                            // global_addr[31:0]
    g0[3] = (unsigned)((ga >> 32) & 0x01ffffffu)     // global_addr[56:32]
          | (2u << 30);                              // type=2
    v8i g1;
    g1[0] = (int)(2u << 16);     // [17:16] data_size = 2 -> 4 bytes
    g1[1] = (int)(256u << 16);   // [63:48]  tensor_dim0 lo16 = 256
    g1[2] = (int)(64u << 16);    // [79:64]  dim0 hi16=0; [95:80] dim1 lo16=64
    g1[3] = (int)(256u << 16);   // [127:112] tile_dim0 = 256 (dim1 hi16=0)
    g1[4] = 64;                  // [143:128] tile_dim1 = 64; tile_dim2 = 0
    g1[5] = 256;                 // [191:160] tensor_dim0_stride lo32 = 256
    g1[6] = 0;                   // stride0 hi16 = 0; stride1 lo16 = 0
    g1[7] = 0;                   // stride1 hi32 = 0
    v4i z4 = {0, 0, 0, 0};               // groups 2/3 unused (2D tensor)
    v8i z8 = {0, 0, 0, 0, 0, 0, 0, 0};   // 6-arg form: extra group, zeroed
    __builtin_amdgcn_tensor_load_to_lds(g0, g1, z4, z4, z8, 0);
    __builtin_amdgcn_s_wait_tensorcnt(0);
  }
  __syncthreads();

  // --- convert fp32 tile -> bf16 fragment tile ---------------------------
#pragma unroll
  for (int i = 0; i < 16; ++i) {
    float4 f = ((const float4*)Afp)[tid * 16 + i];
    As[tid * 32 + 2 * i]     = pack_bf16x2(f.x, f.y);
    As[tid * 32 + 2 * i + 1] = pack_bf16x2(f.z, f.w);
  }
#else
  // Fallback: stage via VGPRs (64 floats/thread).
  {
    const float* Ab = A + row0 * DIMC;
#pragma unroll
    for (int i = 0; i < 16; ++i) {
      float4 f = ((const float4*)Ab)[tid * 16 + i];
      As[tid * 32 + 2 * i]     = pack_bf16x2(f.x, f.y);
      As[tid * 32 + 2 * i + 1] = pack_bf16x2(f.z, f.w);
    }
  }
#endif
  __syncthreads();

  v8f acc[4][2] = {};
  const bf16* Asb = (const bf16*)As;

#pragma unroll
  for (int kt = 0; kt < 8; ++kt) {
    v16bf bfrag[2];
#pragma unroll
    for (int ni = 0; ni < 2; ++ni) {
      int nt = wave * 2 + ni;
      bfrag[ni] = *(const v16bf*)(Wsw + (((nt * 8 + kt) * 32 + lane) << 4));
    }
#pragma unroll
    for (int mi = 0; mi < 4; ++mi) {
      int m = mi * 16 + l15;
      v8bf lo = *(const v8bf*)(Asb + m * DIMC + kt * 32 + half * 16);
      v8bf hi = *(const v8bf*)(Asb + m * DIMC + kt * 32 + half * 16 + 8);
      v16bf af = __builtin_shufflevector(lo, hi, 0, 1, 2, 3, 4, 5, 6, 7,
                                                 8, 9, 10, 11, 12, 13, 14, 15);
#pragma unroll
      for (int ni = 0; ni < 2; ++ni) {
        acc[mi][ni] = __builtin_amdgcn_wmma_f32_16x16x32_bf16(
            false, af, false, bfrag[ni], (short)0, acc[mi][ni], false, false);
      }
    }
  }

  // Epilogue: C/D layout -> lane holds (M = rr + 8*half, N = l15) per VGPR rr.
#pragma unroll
  for (int mi = 0; mi < 4; ++mi) {
#pragma unroll
    for (int ni = 0; ni < 2; ++ni) {
      int col = wave * 32 + ni * 16 + l15;
      float bs = bias[col];
#pragma unroll
      for (int rr = 0; rr < 8; ++rr) {
        size_t r = row0 + mi * 16 + half * 8 + rr;
        float v = acc[mi][ni][rr] + bs;
        if (res)  v += res[r * DIMC + col];
        if (relu) v = fmaxf(v, 0.0f);
        C[r * DIMC + col] = v;
      }
    }
  }
}

// ---------------------------------------------------------------------------
// Score: a[b,m,n,c] = (q[b,m,c] * k[b,n,c] * scale) * (e+1) * e
// One float4 per thread.
// ---------------------------------------------------------------------------
__global__ __launch_bounds__(256) void score_kernel(
    const float* __restrict__ q, const float* __restrict__ k,
    const float* __restrict__ e, float* __restrict__ a, float scale) {
  size_t idx4 = (size_t)blockIdx.x * 256 + threadIdx.x;  // float4 index
  size_t c4 = idx4 & 63;       // 64 float4 per 256-wide row
  size_t r  = idx4 >> 6;       // flattened (b,m,n)
  size_t n  = r & (NSEQ - 1);
  size_t bm = r >> 7;          // b*128 + m
  size_t b  = bm >> 7;
  float4 qv = ((const float4*)q)[bm * 64 + c4];
  float4 kv = ((const float4*)k)[(b * NSEQ + n) * 64 + c4];
  float4 ev = ((const float4*)e)[idx4];
  float4 o;
  o.x = (qv.x * kv.x * scale) * (ev.x + 1.0f) * ev.x;
  o.y = (qv.y * kv.y * scale) * (ev.y + 1.0f) * ev.y;
  o.z = (qv.z * kv.z * scale) * (ev.z + 1.0f) * ev.z;
  o.w = (qv.w * kv.w * scale) * (ev.w + 1.0f) * ev.w;
  ((float4*)a)[idx4] = o;
}

// ---------------------------------------------------------------------------
// Fused softmax(axis=n) + sum_n softmax*V using the online-softmax
// recurrence: one pass over the 128 MB score tensor instead of three.
// Block per (b,m); thread per channel c.
// ---------------------------------------------------------------------------
__global__ __launch_bounds__(256) void softmax_nodes_kernel(
    const float* __restrict__ a, const float* __restrict__ v,
    float* __restrict__ nodes) {
  int bm = blockIdx.x;          // 0..1023
  int c  = threadIdx.x;         // 0..255
  int b  = bm >> 7;
  const float* arow = a + (size_t)bm * NSEQ * DIMC + c;
  const float* vcol = v + (size_t)b * NSEQ * DIMC + c;
  float mx = -3.4e38f, sum = 0.0f, acc = 0.0f;
  for (int n = 0; n < NSEQ; ++n) {
    float x  = arow[(size_t)n * DIMC];
    float vv = vcol[(size_t)n * DIMC];
    float m2 = fmaxf(mx, x);
    float sc = __expf(mx - m2);
    float p  = __expf(x - m2);
    sum = sum * sc + p;
    acc = acc * sc + p * vv;
    mx = m2;
  }
  nodes[(size_t)bm * DIMC + c] = acc / sum;
}

// ---------------------------------------------------------------------------
// Orchestration. Input order (setup_inputs dict order, params depth-first,
// lnp -> {g,b}, lin -> {w,b}):
//  0 mol_annot  1 prot_annot  2 mol_adj  3 prot_adj
//  4..11  ln1_mx{g,b} ln1_px{g,b} ln1_ma{g,b} ln1_pa{g,b}
// 12..23  attn2 q/k/v/e/out_n/out_e {w,b}
// 24..27  ln2_px{g,b} ln2_pa{g,b}
// 28..39  dec q_mx/k_px/v_mx/k_pa/out_nd/out_ed {w,b}
// 40..43  ln3_ma{g,b} ln3_mx{g,b}
// 44..51  mlp_ma fc1/fc2 {w,b}, mlp_mx fc1/fc2 {w,b}
// 52..55  ln4_ma{g,b} ln4_mx{g,b}
// ---------------------------------------------------------------------------
extern "C" void kernel_launch(void* const* d_in, const int* in_sizes, int n_in,
                              void* d_out, int out_size, void* d_ws, size_t ws_size,
                              hipStream_t stream) {
  (void)in_sizes; (void)n_in; (void)out_size; (void)ws_size;

  const float* mol_annot  = (const float*)d_in[0];
  const float* prot_annot = (const float*)d_in[1];
  const float* mol_adj    = (const float*)d_in[2];
  const float* prot_adj   = (const float*)d_in[3];
  auto P = [&](int i) { return (const float*)d_in[i]; };

  // ---- workspace layout -------------------------------------------------
  bf16*  wbf = (bf16*)d_ws;                                  // 16 x 128 KB bf16
  float* fws = (float*)((char*)d_ws + (size_t)16 * 65536 * sizeof(bf16));
  const size_t S = (size_t)RSML * DIMC;                      // 1 MB fp32 each
  auto sm = [&](int i) { return fws + (size_t)i * S; };
  float* bigbase = fws + 16 * S;
  const size_t BIGSZ = (size_t)RBIG * DIMC;                  // 128 MB fp32 each
  auto bg = [&](int i) { return bigbase + (size_t)i * BIGSZ; };

  // ---- pre-swizzle all 16 weight matrices to bf16 fragment order --------
  // slots: 0 q 1 k 2 v 3 e 4 out_n 5 out_e | 6 q_mx 7 k_px 8 v_mx 9 k_pa
  //        10 out_nd 11 out_ed | 12 fc1_ma 13 fc2_ma 14 fc1_mx 15 fc2_mx
  static const int wsrc[16] = {12,14,16,18,20,22, 28,30,32,34,36,38, 44,46,48,50};
  for (int i = 0; i < 16; ++i)
    swizzle_weights<<<256, 256, 0, stream>>>(P(wsrc[i]), wbf + (size_t)i * 65536);
  auto W = [&](int s) { return (const bf16*)(wbf + (size_t)s * 65536); };

  float *mx_ln = sm(0), *px_ln = sm(1);
  float *q1 = sm(2), *k1 = sm(3), *v1 = sm(4);
  float *nodes1 = sm(5), *px1 = sm(6), *px1_ln = sm(7);
  float *q2 = sm(8), *k2 = sm(9), *v2 = sm(10);
  float *nodes2 = sm(11), *mx1 = sm(12), *mx1_ln = sm(13);
  float *hn = sm(14), *mx2 = sm(15);
  float *bigA = bg(0), *bigB = bg(1), *bigC = bg(2), *bigD = bg(3), *bigE = bg(4);

  // ---- initial LayerNorms ----------------------------------------------
  ln_kernel<<<RSML / 8, 256, 0, stream>>>(mol_annot,  P(4),  P(5),  mx_ln);
  ln_kernel<<<RSML / 8, 256, 0, stream>>>(prot_annot, P(6),  P(7),  px_ln);
  ln_kernel<<<RBIG / 8, 256, 0, stream>>>(prot_adj,   P(10), P(11), bigA);   // pa_ln

  // ---- Block 1: protein self edge-attention (attn2) ---------------------
  gemm256<<<RSML / 64, 256, 0, stream>>>(px_ln, W(0), P(13), nullptr, q1, 0);
  gemm256<<<RSML / 64, 256, 0, stream>>>(px_ln, W(1), P(15), nullptr, k1, 0);
  gemm256<<<RSML / 64, 256, 0, stream>>>(px_ln, W(2), P(17), nullptr, v1, 0);
  gemm256<<<RBIG / 64, 256, 0, stream>>>(bigA,  W(3), P(19), nullptr, bigB, 0); // e1
  score_kernel<<<(RBIG * 64) / 256, 256, 0, stream>>>(q1, k1, bigB, bigC,
                                                      0.17677669529663689f); // 1/sqrt(32)
  gemm256<<<RBIG / 64, 256, 0, stream>>>(bigC, W(5), P(23), bigA, bigD, 0);  // pa1 = pa + out_e(a)
  ln_kernel<<<RBIG / 8, 256, 0, stream>>>(bigD, P(26), P(27), bigE);          // pa1_ln
  softmax_nodes_kernel<<<RSML, 256, 0, stream>>>(bigC, v1, nodes1);
  gemm256<<<RSML / 64, 256, 0, stream>>>(nodes1, W(4), P(21), px_ln, px1, 0); // px1 = px + out_n
  ln_kernel<<<RSML / 8, 256, 0, stream>>>(px1, P(24), P(25), px1_ln);

  // ---- Block 2: decoder cross edge-attention ---------------------------
  ln_kernel<<<RBIG / 8, 256, 0, stream>>>(mol_adj, P(8), P(9), bigA);         // ma_ln
  gemm256<<<RSML / 64, 256, 0, stream>>>(px1_ln, W(6), P(29), nullptr, q2, 0);
  gemm256<<<RSML / 64, 256, 0, stream>>>(mx_ln,  W(7), P(31), nullptr, k2, 0);
  gemm256<<<RSML / 64, 256, 0, stream>>>(px1_ln, W(8), P(33), nullptr, v2, 0);
  gemm256<<<RBIG / 64, 256, 0, stream>>>(bigE, W(9), P(35), nullptr, bigB, 0); // e2 from pa1_ln
  score_kernel<<<(RBIG * 64) / 256, 256, 0, stream>>>(q2, k2, bigB, bigC,
                                                      0.0625f);               // 1/sqrt(256)
  gemm256<<<RBIG / 64, 256, 0, stream>>>(bigC, W(11), P(39), bigA, bigD, 0);  // ma1 = ma + out_ed(a)
  ln_kernel<<<RBIG / 8, 256, 0, stream>>>(bigD, P(40), P(41), bigE);          // ma1_ln
  softmax_nodes_kernel<<<RSML, 256, 0, stream>>>(bigC, v2, nodes2);
  gemm256<<<RSML / 64, 256, 0, stream>>>(nodes2, W(10), P(37), mx_ln, mx1, 0);
  ln_kernel<<<RSML / 8, 256, 0, stream>>>(mx1, P(42), P(43), mx1_ln);

  // ---- MLPs + final LayerNorms -----------------------------------------
  float* out_edge = (float*)d_out;
  float* out_node = out_edge + BIGSZ;
  gemm256<<<RBIG / 64, 256, 0, stream>>>(bigE, W(12), P(45), nullptr, bigA, 1); // relu(fc1)
  gemm256<<<RBIG / 64, 256, 0, stream>>>(bigA, W(13), P(47), bigE, bigB, 0);    // fc2 + ma1_ln
  ln_kernel<<<RBIG / 8, 256, 0, stream>>>(bigB, P(52), P(53), out_edge);        // edge_hidden

  gemm256<<<RSML / 64, 256, 0, stream>>>(mx1_ln, W(14), P(49), nullptr, hn, 1);
  gemm256<<<RSML / 64, 256, 0, stream>>>(hn, W(15), P(51), mx1_ln, mx2, 0);
  ln_kernel<<<RSML / 8, 256, 0, stream>>>(mx2, P(54), P(55), out_node);         // node_hidden
}
